// AudioModelX3_88424786690501
// MI455X (gfx1250) — compile-verified, matrix operand order checked by reading, not run on hardware
//
#include <hip/hip_runtime.h>
#include <math.h>

// ---------------- model constants ----------------
constexpr int Bc   = 4;
constexpr int Sc   = 1024;
constexpr int Dc   = 1024;
constexpr int NN   = Bc * Sc;        // 4096 rows
constexpr int INNERc = 2048;
constexpr int NHm  = 4;
constexpr int DHm  = 512;
constexpr int NHs  = 4;
constexpr int DHs  = 256;
constexpr int FFUP = 1344;

typedef __attribute__((ext_vector_type(16))) __bf16 v16bf;
typedef __attribute__((ext_vector_type(8)))  __bf16 v8bf;
typedef __attribute__((ext_vector_type(4)))  __bf16 v4bf;
typedef __attribute__((ext_vector_type(8)))  float  v8f;

static __device__ __forceinline__ __bf16 f2bf(float x) { return (__bf16)x; }

static __device__ __forceinline__ v16bf cat16(v8bf lo, v8bf hi) {
  return __builtin_shufflevector(lo, hi, 0, 1, 2, 3, 4, 5, 6, 7,
                                 8, 9, 10, 11, 12, 13, 14, 15);
}

// swizzled-LDS coordinates for a K index within a 32-chunk (CDNA5 ISA 7.12.2):
// lane<16 holds K {0..7,16..23}; lane>=16 holds K {8..15,24..31}.
static __device__ __forceinline__ int kc_h(int kc) { return (kc >> 3) & 1; }
static __device__ __forceinline__ int kc_e(int kc) { return (kc & 7) + ((kc >> 4) & 1) * 8; }

static __device__ __forceinline__ float logsigf(float x) {
  return fminf(x, 0.f) - log1pf(__expf(-fabsf(x)));
}
static __device__ __forceinline__ float siluf(float x) {
  return x / (1.f + __expf(-x));
}
static __device__ __forceinline__ float geluf(float x) { // tanh approx (jax default)
  float x3 = x * x * x;
  return 0.5f * x * (1.f + tanhf(0.7978845608028654f * (x + 0.044715f * x3)));
}

// =====================================================================
// Weight prep: fp32 (Kd x M) row-major  ->  bf16 (M x Kd) row-major
// =====================================================================
__global__ __launch_bounds__(256) void wconvT_kernel(
    const float* __restrict__ W, __bf16* __restrict__ Wt, int Kd, int M)
{
  size_t idx = (size_t)blockIdx.x * 256 + threadIdx.x;
  if (idx >= (size_t)Kd * M) return;
  int col = (int)(idx / Kd), k = (int)(idx % Kd);
  Wt[idx] = f2bf(W[(size_t)k * M + col]);
}

// Weight prep: straight fp32 -> bf16 copy (layout already M x Kd)
__global__ __launch_bounds__(256) void wconv_kernel(
    const float* __restrict__ W, __bf16* __restrict__ Wt, int n)
{
  size_t idx = (size_t)blockIdx.x * 256 + threadIdx.x;
  if (idx < (size_t)n) Wt[idx] = f2bf(W[idx]);
}

// q/k bf16 prep (q pre-scaled by DH^-0.5)
__global__ __launch_bounds__(256) void qkbf_kernel(
    const float* __restrict__ q, const float* __restrict__ k,
    __bf16* __restrict__ qb, __bf16* __restrict__ kb)
{
  size_t idx = (size_t)blockIdx.x * 256 + threadIdx.x;
  if (idx >= (size_t)NN * INNERc) return;
  qb[idx] = f2bf(q[idx] * 0.044194173824159216f);
  kb[idx] = f2bf(k[idx]);
}

// V transpose prep: Vt[((b*NH+h)*DH + col)*S + key] = bf16(v[(b*S+key)*INNER + h*DH + col])
__global__ __launch_bounds__(256) void vt_kernel(
    const float* __restrict__ v, __bf16* __restrict__ vt)
{
  size_t idx = (size_t)blockIdx.x * 256 + threadIdx.x;
  if (idx >= (size_t)NN * INNERc) return;
  int key = (int)(idx & 1023);
  int col = (int)((idx >> 10) & 511);
  int h   = (int)((idx >> 19) & 3);
  int b   = (int)(idx >> 21);
  vt[idx] = f2bf(v[((size_t)(b * Sc + key)) * INNERc + h * DHm + col]);
}

// =====================================================================
// GEMM:  C[N x M] = (res?) + A[N x Kd] @ W,  W pre-converted bf16 (M x Kd).
// block = 128 threads (4 waves); tile 64x32; K step 32; bf16 WMMA f32-acc.
// A staged through swizzled LDS (b128 in, packed cvt, b128 out);
// B fragments read directly from L2-resident bf16 weights (2x b128/lane).
// =====================================================================
__global__ __launch_bounds__(128) void gemm_bf16_wmma(
    const float* __restrict__ A, int lda,
    const __bf16* __restrict__ Wt,
    const float* __restrict__ res,
    float* __restrict__ C, int ldc,
    int N, int Kd, int M)
{
  __shared__ __align__(16) __bf16 As[64 * 40];

  const int tid  = threadIdx.x;
  const int w    = tid >> 5;
  const int lane = tid & 31;
  const int rl   = lane & 15;
  const int hhf  = (lane >> 4) & 1;
  const int klo  = hhf * 8;
  const int rowbase = blockIdx.x * 64;
  const int colbase = blockIdx.y * 32;

  v8f acc[2];
#pragma unroll
  for (int nt = 0; nt < 2; ++nt)
#pragma unroll
    for (int i = 0; i < 8; ++i) acc[nt][i] = 0.f;

  const int arow = rowbase + (tid >> 1);      // this thread's A row
  const int ahf  = (tid & 1) * 16;            // k-subrange [ahf, ahf+16)
  const float* Arow = A + (size_t)arow * lda + ahf;
  const __bf16* Wc0 = Wt + (size_t)(colbase + rl) * Kd;
  const __bf16* Wc1 = Wt + (size_t)(colbase + 16 + rl) * Kd;

  for (int kk = 0; kk < Kd; kk += 32) {
    // ---- A tile: 16 consecutive k per thread as 4x float4 -> swizzled bf16
#pragma unroll
    for (int g = 0; g < 4; ++g) {
      float4 v4 = *(const float4*)(Arow + kk + g * 4);
      int k0 = ahf + g * 4;
      v4bf pk;
      pk[0] = f2bf(v4.x); pk[1] = f2bf(v4.y); pk[2] = f2bf(v4.z); pk[3] = f2bf(v4.w);
      *(v4bf*)&As[(tid >> 1) * 40 + kc_h(k0) * 16 + kc_e(k0)] = pk;
    }
    if (kk + 32 < Kd)  // emits global_prefetch_b8
      __builtin_prefetch(Arow + kk + 32, 0, 0);
    __syncthreads();

    v16bf af = *(const v16bf*)&As[(w * 16 + rl) * 40 + hhf * 16];
    {
      v16bf b0 = cat16(*(const v8bf*)(Wc0 + kk + klo),
                       *(const v8bf*)(Wc0 + kk + 16 + klo));
      acc[0] = __builtin_amdgcn_wmma_f32_16x16x32_bf16(false, af, false, b0,
                                                       (short)0, acc[0], false, false);
      v16bf b1 = cat16(*(const v8bf*)(Wc1 + kk + klo),
                       *(const v8bf*)(Wc1 + kk + 16 + klo));
      acc[1] = __builtin_amdgcn_wmma_f32_16x16x32_bf16(false, af, false, b1,
                                                       (short)0, acc[1], false, false);
    }
    __syncthreads();
  }

  const int rhi = hhf * 8;
#pragma unroll
  for (int nt = 0; nt < 2; ++nt) {
#pragma unroll
    for (int g = 0; g < 8; ++g) {
      int grow = rowbase + w * 16 + rhi + g;
      int gcol = colbase + nt * 16 + rl;
      float v_ = acc[nt][g];
      if (res) v_ += res[(size_t)grow * ldc + gcol];
      C[(size_t)grow * ldc + gcol] = v_;
    }
  }
}

// =====================================================================
// Row LayerNorm (optionally SELU epilogue). One block per row.
// =====================================================================
__global__ __launch_bounds__(256) void ln_kernel(
    const float* __restrict__ in, const float* __restrict__ w,
    float* __restrict__ out, int C, int doSelu)
{
  __shared__ float s1[256], s2[256];
  const int row = blockIdx.x, tid = threadIdx.x;
  const size_t base = (size_t)row * C;
  float a = 0.f, b = 0.f;
  for (int c = tid; c < C; c += 256) { float x = in[base + c]; a += x; b += x * x; }
  s1[tid] = a; s2[tid] = b; __syncthreads();
  for (int off = 128; off; off >>= 1) {
    if (tid < off) { s1[tid] += s1[tid + off]; s2[tid] += s2[tid + off]; }
    __syncthreads();
  }
  const float mean = s1[0] / C;
  const float var  = s2[0] / C - mean * mean;
  const float rstd = rsqrtf(var + 1e-5f);
  for (int c = tid; c < C; c += 256) {
    float y = (in[base + c] - mean) * rstd * w[c];
    if (doSelu)
      y = 1.0507009873554805f * (y > 0.f ? y : 1.6732632423543772f * (__expf(y) - 1.f));
    out[base + c] = y;
  }
}

// =====================================================================
// Multi-head LayerNorm: rows are (n*NH + hd), length C=DH, weight w[hd*C+c].
// Optional residual add. Strided in/out (row n stride ldin/ldout).
// =====================================================================
__global__ __launch_bounds__(256) void mhln_kernel(
    const float* __restrict__ in, int ldin, const float* __restrict__ w,
    float* __restrict__ out, int ldout, const float* __restrict__ res,
    int C, int NH)
{
  __shared__ float s1[256], s2[256];
  const int row = blockIdx.x, tid = threadIdx.x;
  const int n = row / NH, hd = row % NH;
  const size_t bin  = (size_t)n * ldin  + hd * C;
  const size_t bout = (size_t)n * ldout + hd * C;
  float a = 0.f, b = 0.f;
  for (int c = tid; c < C; c += 256) { float x = in[bin + c]; a += x; b += x * x; }
  s1[tid] = a; s2[tid] = b; __syncthreads();
  for (int off = 128; off; off >>= 1) {
    if (tid < off) { s1[tid] += s1[tid + off]; s2[tid] += s2[tid + off]; }
    __syncthreads();
  }
  const float mean = s1[0] / C;
  const float var  = s2[0] / C - mean * mean;
  const float rstd = rsqrtf(var + 1e-5f);
  for (int c = tid; c < C; c += 256) {
    float y = (in[bin + c] - mean) * rstd * w[hd * C + c];
    if (res) y += res[bout + c];
    out[bout + c] = y;
  }
}

// ===================== depthwise causal conv (K=4) + SiLU =====================
__global__ __launch_bounds__(256) void conv_silu_kernel(
    const float* __restrict__ x, int ldx,
    const float* __restrict__ w, const float* __restrict__ bias,
    float* __restrict__ y, int ldy, int C)
{
  size_t idx = (size_t)blockIdx.x * 256 + threadIdx.x;
  if (idx >= (size_t)NN * C) return;
  int n = (int)(idx / C), c = (int)(idx % C);
  int s = n % Sc;
  float acc = bias[c];
  float4 wv = *(const float4*)&w[c * 4];
  const float wk[4] = {wv.x, wv.y, wv.z, wv.w};
#pragma unroll
  for (int t = 0; t < 4; ++t) {
    int d = 3 - t;
    if (s - d >= 0) acc += x[(size_t)(n - d) * ldx + c] * wk[t];
  }
  y[(size_t)n * ldy + c] = siluf(acc);
}

// ===================== 4x4 block-diagonal projection =====================
__global__ __launch_bounds__(256) void headwise4_kernel(
    const float* __restrict__ x, int ldx, const float* __restrict__ W,
    float* __restrict__ y, int ldy, int C)
{
  size_t idx = (size_t)blockIdx.x * 256 + threadIdx.x;
  if (idx >= (size_t)NN * C) return;
  int n = (int)(idx / C), c = (int)(idx % C);
  int g = c >> 2, o = c & 3;
  float4 xb = *(const float4*)&x[(size_t)n * ldx + g * 4];
  float4 wb = *(const float4*)&W[((size_t)g * 4 + o) * 4];
  y[(size_t)n * ldy + c] = xb.x*wb.x + xb.y*wb.y + xb.z*wb.z + xb.w*wb.w;
}

// ===================== mLSTM gate pre-activations (i/f, NH=4 each) ============
__global__ __launch_bounds__(256) void gates_kernel(
    const float* __restrict__ q, const float* __restrict__ k,
    const float* __restrict__ v,
    const float* __restrict__ Wig, const float* __restrict__ big,
    const float* __restrict__ Wfg, const float* __restrict__ bfg,
    float* __restrict__ ig, float* __restrict__ fg)
{
  __shared__ float red[8][256];
  const int n = blockIdx.x, tid = threadIdx.x;
  float acc[8];
#pragma unroll
  for (int g = 0; g < 8; ++g) acc[g] = 0.f;
  const size_t base = (size_t)n * INNERc;
  for (int j = tid; j < INNERc; j += 256) {
    float xv[3] = {q[base + j], k[base + j], v[base + j]};
    int jj[3] = {j, INNERc + j, 2 * INNERc + j};
#pragma unroll
    for (int t = 0; t < 3; ++t) {
      float4 wi = *(const float4*)&Wig[jj[t] * 4];
      float4 wf = *(const float4*)&Wfg[jj[t] * 4];
      acc[0] += xv[t] * wi.x; acc[1] += xv[t] * wi.y;
      acc[2] += xv[t] * wi.z; acc[3] += xv[t] * wi.w;
      acc[4] += xv[t] * wf.x; acc[5] += xv[t] * wf.y;
      acc[6] += xv[t] * wf.z; acc[7] += xv[t] * wf.w;
    }
  }
#pragma unroll
  for (int g = 0; g < 8; ++g) red[g][tid] = acc[g];
  __syncthreads();
  for (int off = 128; off; off >>= 1) {
    if (tid < off)
#pragma unroll
      for (int g = 0; g < 8; ++g) red[g][tid] += red[g][tid + off];
    __syncthreads();
  }
  if (tid < 4)      ig[(size_t)n * 4 + tid] = red[tid][0]     + big[tid];
  else if (tid < 8) fg[(size_t)n * 4 + tid - 4] = red[tid][0] + bfg[tid - 4];
}

// ===================== cumulative log-sigmoid of forget gates =====================
__global__ void lfc_kernel(const float* __restrict__ fg, float* __restrict__ lfc)
{
  if (threadIdx.x != 0) return;
  const int b = blockIdx.x >> 2, h = blockIdx.x & 3;
  float acc = 0.f;
  for (int s = 0; s < Sc; ++s) {
    acc += logsigf(fg[((size_t)(b * Sc + s)) * 4 + h]);
    lfc[(size_t)(b * 4 + h) * Sc + s] = acc;
  }
}

// =====================================================================
// Flash-style mLSTM attention. Block = 128 threads (4 waves).
// Block handles (b, head, 16-query tile); wave w owns output cols [w*128,+128).
// All WMMA operands read directly from bf16 global (2x global_load_b128/lane):
// q fragments hoisted to registers, k streamed, V pre-transposed.
// Online max/sum stabilizer; LDS only for score-partials + P + row stats.
// =====================================================================
__global__ __launch_bounds__(128) void mlstm_attn_kernel(
    const __bf16* __restrict__ Qb, const __bf16* __restrict__ Kb,
    const __bf16* __restrict__ Vt, const float* __restrict__ IG,
    const float* __restrict__ LFC, float* __restrict__ H, int ldh)
{
  __shared__ float  Sp[4][16][32];
  __shared__ __align__(16) __bf16 Pb[16 * 40];
  __shared__ float  lq[16], mrow[16], srow[16], sscale[16], sinv[16];

  const int bi = blockIdx.x, hh = blockIdx.y, b = blockIdx.z;
  const int tid = threadIdx.x, w = tid >> 5, lane = tid & 31;
  const int rl = lane & 15, hhf = (lane >> 4) & 1, rhi = hhf * 8, klo = hhf * 8;

  // ---- hoist the 4 q fragments this wave uses (dims chunks 4w..4w+3)
  v16bf qf[4];
  {
    const __bf16* qp = Qb + ((size_t)(b * Sc + bi * 16 + rl)) * INNERc + hh * DHm;
#pragma unroll
    for (int cc = 0; cc < 4; ++cc) {
      int co = (w * 4 + cc) * 32 + klo;
      qf[cc] = cat16(*(const v8bf*)(qp + co), *(const v8bf*)(qp + co + 16));
    }
  }
  if (tid < 16) {
    lq[tid]   = LFC[(size_t)(b * 4 + hh) * Sc + bi * 16 + tid];
    mrow[tid] = -__builtin_inff();
    srow[tid] = 0.f;
  }

  v8f oacc[8];
#pragma unroll
  for (int nt = 0; nt < 8; ++nt)
#pragma unroll
    for (int g = 0; g < 8; ++g) oacc[nt][g] = 0.f;

  const __bf16* Vh = Vt + (size_t)(b * 4 + hh) * DHm * Sc;
  const int jtmax = (bi * 16 + 15) >> 5;
  for (int jt = 0; jt <= jtmax; ++jt) {
    // ---- scores: wave w handles dim-chunks [4w, 4w+4) for both key halves
    const __bf16* kp0 = Kb + ((size_t)(b * Sc + jt * 32 + rl)) * INNERc + hh * DHm;
    const __bf16* kp1 = kp0 + (size_t)16 * INNERc;
    v8f sa0, sa1;
#pragma unroll
    for (int g = 0; g < 8; ++g) { sa0[g] = 0.f; sa1[g] = 0.f; }
#pragma unroll
    for (int cc = 0; cc < 4; ++cc) {
      int co = (w * 4 + cc) * 32 + klo;
      v16bf b0 = cat16(*(const v8bf*)(kp0 + co), *(const v8bf*)(kp0 + co + 16));
      v16bf b1 = cat16(*(const v8bf*)(kp1 + co), *(const v8bf*)(kp1 + co + 16));
      sa0 = __builtin_amdgcn_wmma_f32_16x16x32_bf16(false, qf[cc], false, b0, (short)0, sa0, false, false);
      sa1 = __builtin_amdgcn_wmma_f32_16x16x32_bf16(false, qf[cc], false, b1, (short)0, sa1, false, false);
    }
#pragma unroll
    for (int g = 0; g < 8; ++g) {
      Sp[w][rhi + g][rl]      = sa0[g];
      Sp[w][rhi + g][16 + rl] = sa1[g];
    }
    __syncthreads();

    // ---- online mLSTM stabilizer + weighted P (threads 0..15, one row each)
    if (tid < 16) {
      const int r = tid, qrow = bi * 16 + r;
      float sv[32], ldv[32];
      float tm = -__builtin_inff();
#pragma unroll
      for (int c2 = 0; c2 < 32; ++c2) {
        sv[c2] = Sp[0][r][c2] + Sp[1][r][c2] + Sp[2][r][c2] + Sp[3][r][c2];
        int kkg = jt * 32 + c2;
        if (kkg <= qrow) {
          float ld_ = lq[r] - LFC[(size_t)(b * 4 + hh) * Sc + kkg]
                            + IG[((size_t)(b * Sc + kkg)) * 4 + hh];
          ldv[c2] = ld_;
          tm = fmaxf(tm, ld_);
        } else ldv[c2] = 0.f;
      }
      float nm = fmaxf(mrow[r], tm);
      float sc = __expf(mrow[r] - nm);
      float ss = srow[r] * sc;
#pragma unroll
      for (int c2 = 0; c2 < 32; ++c2) {
        int kkg = jt * 32 + c2;
        float p = 0.f;
        if (kkg <= qrow) p = sv[c2] * __expf(ldv[c2] - nm);
        Pb[r * 40 + kc_h(c2) * 16 + kc_e(c2)] = f2bf(p);
        ss += p;
      }
      srow[r] = ss; mrow[r] = nm; sscale[r] = sc;
    }
    __syncthreads();

    // ---- rescale accumulators, then oacc += P @ Vtile (V fragments from Vt)
#pragma unroll
    for (int g = 0; g < 8; ++g) {
      float sc = sscale[rhi + g];
#pragma unroll
      for (int nt = 0; nt < 8; ++nt) oacc[nt][g] *= sc;
    }
    v16bf pf = *(const v16bf*)&Pb[rl * 40 + hhf * 16];
#pragma unroll
    for (int nt = 0; nt < 8; ++nt) {
      int colg = w * 128 + nt * 16 + rl;
      const __bf16* vp = Vh + (size_t)colg * Sc + jt * 32 + klo;
      v16bf vf = cat16(*(const v8bf*)vp, *(const v8bf*)(vp + 16));
      oacc[nt] = __builtin_amdgcn_wmma_f32_16x16x32_bf16(false, pf, false, vf, (short)0, oacc[nt], false, false);
    }
  }

  __syncthreads();
  if (tid < 16) {
    float nr = fmaxf(fabsf(srow[tid]), __expf(-mrow[tid])) + 1e-6f;
    sinv[tid] = 1.f / nr;
  }
  __syncthreads();
#pragma unroll
  for (int g = 0; g < 8; ++g) {
    int rlo = rhi + g;
    float inv = sinv[rlo];
    int qrow = bi * 16 + rlo;
#pragma unroll
    for (int nt = 0; nt < 8; ++nt) {
      int colg = w * 128 + nt * 16 + rl;
      H[((size_t)(b * Sc + qrow)) * ldh + hh * DHm + colg] = oacc[nt][g] * inv;
    }
  }
}

// ===================== mLSTM gated-combine: G = (hn + skip*xc) * silu(z) ========
__global__ __launch_bounds__(256) void gcombine_kernel(
    const float* __restrict__ hn, const float* __restrict__ skip,
    const float* __restrict__ xc, const float* __restrict__ up,
    float* __restrict__ G)
{
  size_t idx = (size_t)blockIdx.x * 256 + threadIdx.x;
  if (idx >= (size_t)NN * INNERc) return;
  int n = (int)(idx / INNERc), c = (int)(idx % INNERc);
  float z = up[(size_t)n * 4096 + 2048 + c];
  G[idx] = (hn[idx] + skip[c] * xc[idx]) * siluf(z);
}

// ===================== repack sLSTM recurrent matrix: R4[h][dd][d][4] ==========
__global__ __launch_bounds__(256) void r4_pack_kernel(
    const float* __restrict__ R, float* __restrict__ R4)
{
  size_t idx = (size_t)blockIdx.x * 256 + threadIdx.x;
  if (idx >= (size_t)NHs * DHs * DHs * 4) return;
  int g  = (int)(idx & 3);
  int d  = (int)((idx >> 2) & 255);
  int dd = (int)((idx >> 10) & 255);
  int h  = (int)(idx >> 18);
  R4[idx] = R[((size_t)h * DHs + dd) * (4 * DHs) + g * DHs + d];
}

// ===================== sLSTM sequential scan (block per (b,head)) ==============
__global__ __launch_bounds__(256) void slstm_kernel(
    const float* __restrict__ ip, const float* __restrict__ fp,
    const float* __restrict__ zp, const float* __restrict__ op,
    const float* __restrict__ R4, const float* __restrict__ bias,
    float* __restrict__ hs)
{
  __shared__ float hsh[256];
  const int b = blockIdx.x >> 2, hh = blockIdx.x & 3;
  const int d = threadIdx.x;
  const float4* Rh = (const float4*)(R4 + (size_t)hh * DHs * DHs * 4);
  const float bi_ = bias[(hh * 4 + 0) * DHs + d];
  const float bf_ = bias[(hh * 4 + 1) * DHs + d];
  const float bz_ = bias[(hh * 4 + 2) * DHs + d];
  const float bo_ = bias[(hh * 4 + 3) * DHs + d];
  float c = 0.f, n = 0.f, m = 0.f;
  hsh[d] = 0.f;
  __syncthreads();
  for (int s = 0; s < Sc; ++s) {
    const size_t nb = (size_t)(b * Sc + s) * Dc + hh * DHs + d;
    float ix = ip[nb] + bi_, fx = fp[nb] + bf_, zx = zp[nb] + bz_, ox = op[nb] + bo_;
    float r0 = 0.f, r1 = 0.f, r2 = 0.f, r3 = 0.f;
#pragma unroll 4
    for (int dd = 0; dd < DHs; ++dd) {
      float hv = hsh[dd];
      float4 rv = Rh[dd * DHs + d];
      r0 += hv * rv.x; r1 += hv * rv.y; r2 += hv * rv.z; r3 += hv * rv.w;
    }
    float ir = ix + r0, fr = fx + r1, zr = zx + r2, og = ox + r3;
    float lfm = m + logsigf(fr);
    float mn = fmaxf(ir, lfm);
    float ig_ = __expf(ir - mn), fg_ = __expf(lfm - mn);
    c = fg_ * c + ig_ * tanhf(zr);
    n = fg_ * n + ig_;
    m = mn;
    float hnew = (1.f / (1.f + __expf(-og))) * (c / (n + 1e-6f));
    __syncthreads();
    hsh[d] = hnew;
    __syncthreads();
    hs[nb] = hnew;
  }
}

// ===================== FFN gate: gu = gelu(g) * u ==============================
__global__ __launch_bounds__(256) void gelu_mul_kernel(
    const float* __restrict__ up, float* __restrict__ gu)
{
  size_t idx = (size_t)blockIdx.x * 256 + threadIdx.x;
  if (idx >= (size_t)NN * FFUP) return;
  int n = (int)(idx / FFUP), c = (int)(idx % FFUP);
  float g = up[(size_t)n * (2 * FFUP) + c];
  float u = up[(size_t)n * (2 * FFUP) + FFUP + c];
  gu[idx] = geluf(g) * u;
}

// ===================== mean pool over sequence ================================
__global__ __launch_bounds__(256) void pool_kernel(
    const float* __restrict__ xf, float* __restrict__ feat)
{
  const int b = blockIdx.y;
  const int c = blockIdx.x * 256 + threadIdx.x;
  float acc = 0.f;
  for (int s = 0; s < Sc; ++s) acc += xf[(size_t)(b * Sc + s) * Dc + c];
  feat[(size_t)b * Dc + c] = acc * (1.f / Sc);
}

// ===================== classification heads ===================================
__global__ __launch_bounds__(256) void heads_kernel(
    const float* __restrict__ feat,
    const float* __restrict__ We, const float* __restrict__ be,
    const float* __restrict__ Ws, const float* __restrict__ bs,
    float* __restrict__ out)
{
  __shared__ float red[256];
  const int o = blockIdx.x, tid = threadIdx.x;
  const int b = o / 10, j = o % 10;
  const float* W; float bias; int ow, col;
  if (j < 7) { W = We; bias = be[j];     ow = 7; col = j; }
  else       { W = Ws; bias = bs[j - 7]; ow = 3; col = j - 7; }
  float acc = 0.f;
  for (int d = tid; d < Dc; d += 256)
    acc += feat[(size_t)b * Dc + d] * W[(size_t)d * ow + col];
  red[tid] = acc; __syncthreads();
  for (int off = 128; off; off >>= 1) {
    if (tid < off) red[tid] += red[tid + off];
    __syncthreads();
  }
  if (tid == 0) out[b * 10 + j] = red[0] + bias;
}

// =====================================================================
extern "C" void kernel_launch(void* const* d_in, const int* in_sizes, int n_in,
                              void* d_out, int out_size, void* d_ws, size_t ws_size,
                              hipStream_t stream)
{
  (void)in_sizes; (void)n_in; (void)out_size; (void)ws_size;

  const float* X       = (const float*)d_in[0];
  const float* m_ln_w  = (const float*)d_in[1];
  const float* m_Wup   = (const float*)d_in[2];
  const float* m_cw    = (const float*)d_in[3];
  const float* m_cb    = (const float*)d_in[4];
  const float* m_Wq    = (const float*)d_in[5];
  const float* m_Wk    = (const float*)d_in[6];
  const float* m_Wv    = (const float*)d_in[7];
  const float* m_Wig   = (const float*)d_in[8];
  const float* m_big   = (const float*)d_in[9];
  const float* m_Wfg   = (const float*)d_in[10];
  const float* m_bfg   = (const float*)d_in[11];
  const float* m_mhln  = (const float*)d_in[12];
  const float* m_skip  = (const float*)d_in[13];
  const float* m_Wdown = (const float*)d_in[14];
  const float* s_ln_w  = (const float*)d_in[15];
  const float* s_cw    = (const float*)d_in[16];
  const float* s_cb    = (const float*)d_in[17];
  const float* s_Wi    = (const float*)d_in[18];
  const float* s_Wf    = (const float*)d_in[19];
  const float* s_Wz    = (const float*)d_in[20];
  const float* s_Wo    = (const float*)d_in[21];
  const float* s_R     = (const float*)d_in[22];
  const float* s_bb    = (const float*)d_in[23];
  const float* s_mhln  = (const float*)d_in[24];
  const float* s_ffnln = (const float*)d_in[25];
  const float* s_Wup   = (const float*)d_in[26];
  const float* s_Wdn2  = (const float*)d_in[27];
  const float* post_ln = (const float*)d_in[28];
  const float* h_We    = (const float*)d_in[29];
  const float* h_be    = (const float*)d_in[30];
  const float* h_Ws    = (const float*)d_in[31];
  const float* h_bs    = (const float*)d_in[32];
  float* OUT = (float*)d_out;

  // ---- workspace layout (floats); ~332 MB required ----
  float* ws = (float*)d_ws;
  constexpr size_t Mf = 1u << 20;
  float* Abuf = ws;               // 4M  : xn / xn2 / xn3 / xf
  float* UP   = ws + 4  * Mf;     // 16M : m up (xi|z, ld 4096) -> h (strided) -> gate preacts -> ffn up
  float* XCb  = ws + 20 * Mf;     // 8M  : xc -> xc2
  float* Qb   = ws + 28 * Mf;     // 8M  : q -> hn -> gu
  float* Kb   = ws + 36 * Mf;     // 8M  : k -> G -> hs
  float* Vb   = ws + 44 * Mf;     // 8M  : v
  float* X1   = ws + 52 * Mf;     // 4M  : residual stream 1 / x3
  float* X2   = ws + 56 * Mf;     // 4M  : residual stream 2
  float* IGb  = ws + 60 * Mf;
  float* FGb  = IGb + 16384;
  float* LFCb = FGb + 16384;
  float* FEAT = LFCb + 16384;
  float* R4   = ws + 61 * Mf;     // 4M  : repacked sLSTM recurrent matrix
  __bf16* Qbf = (__bf16*)(ws + 65 * Mf);   // 8M bf16
  __bf16* Kbf = (__bf16*)(ws + 69 * Mf);   // 8M bf16
  __bf16* Vtb = (__bf16*)(ws + 73 * Mf);   // 8M bf16 (transposed V)
  __bf16* WBF = (__bf16*)(ws + 77 * Mf);   // bf16 weight pool (~11M bf16)

  // bf16 weight pool offsets (elements)
  size_t oW = 0;
  __bf16* WupT  = WBF + oW; oW += (size_t)(2 * INNERc) * Dc;      // 4096x1024
  __bf16* WdnT  = WBF + oW; oW += (size_t)Dc * INNERc;            // 1024x2048
  __bf16* WgT   = WBF + oW; oW += (size_t)4 * NHs * DHs * DHs;    // 4 x (4x256x256)
  __bf16* Wup2T = WBF + oW; oW += (size_t)(2 * FFUP) * Dc;        // 2688x1024
  __bf16* Wdn2T = WBF + oW; oW += (size_t)Dc * FFUP;              // 1024x1344

  auto gemm = [&](const float* A, int lda, const __bf16* Wt,
                  const float* res, float* C, int ldc, int N_, int Kd, int M_) {
    dim3 g(N_ / 64, M_ / 32);
    gemm_bf16_wmma<<<g, 128, 0, stream>>>(A, lda, Wt, res, C, ldc, N_, Kd, M_);
  };

  // ================= weight prep (bf16, M x K) =================
  wconvT_kernel<<<((2 * INNERc) * Dc) / 256, 256, 0, stream>>>(m_Wup, WupT, Dc, 2 * INNERc);
  wconvT_kernel<<<(Dc * INNERc) / 256, 256, 0, stream>>>(m_Wdown, WdnT, INNERc, Dc);
  {
    const float* Wg[4] = {s_Wi, s_Wf, s_Wz, s_Wo};
    for (int g = 0; g < 4; ++g)
      wconv_kernel<<<(NHs * DHs * DHs) / 256, 256, 0, stream>>>(
          Wg[g], WgT + (size_t)g * NHs * DHs * DHs, NHs * DHs * DHs);
  }
  wconvT_kernel<<<((2 * FFUP) * Dc) / 256, 256, 0, stream>>>(s_Wup, Wup2T, Dc, 2 * FFUP);
  wconvT_kernel<<<(Dc * FFUP) / 256, 256, 0, stream>>>(s_Wdn2, Wdn2T, FFUP, Dc);
  r4_pack_kernel<<<(NHs * DHs * DHs * 4) / 256, 256, 0, stream>>>(s_R, R4);

  // ================= block 0: mLSTM =================
  ln_kernel<<<NN, 256, 0, stream>>>(X, m_ln_w, Abuf, Dc, 0);
  gemm(Abuf, Dc, WupT, nullptr, UP, 2 * INNERc, NN, Dc, 2 * INNERc);
  conv_silu_kernel<<<(NN * INNERc) / 256, 256, 0, stream>>>(UP, 4096, m_cw, m_cb, XCb, INNERc, INNERc);
  headwise4_kernel<<<(NN * INNERc) / 256, 256, 0, stream>>>(XCb, INNERc, m_Wq, Qb, INNERc, INNERc);
  headwise4_kernel<<<(NN * INNERc) / 256, 256, 0, stream>>>(XCb, INNERc, m_Wk, Kb, INNERc, INNERc);
  headwise4_kernel<<<(NN * INNERc) / 256, 256, 0, stream>>>(UP, 4096, m_Wv, Vb, INNERc, INNERc);
  gates_kernel<<<NN, 256, 0, stream>>>(Qb, Kb, Vb, m_Wig, m_big, m_Wfg, m_bfg, IGb, FGb);
  lfc_kernel<<<16, 32, 0, stream>>>(FGb, LFCb);
  qkbf_kernel<<<(NN * INNERc) / 256, 256, 0, stream>>>(Qb, Kb, Qbf, Kbf);
  vt_kernel<<<(NN * INNERc) / 256, 256, 0, stream>>>(Vb, Vtb);
  {
    dim3 g(Sc / 16, NHm, Bc);
    mlstm_attn_kernel<<<g, 128, 0, stream>>>(Qbf, Kbf, Vtb, IGb, LFCb, UP, 4096);
  }
  mhln_kernel<<<NN * NHm, 256, 0, stream>>>(UP, 4096, m_mhln, Qb, INNERc, nullptr, DHm, NHm);
  gcombine_kernel<<<(NN * INNERc) / 256, 256, 0, stream>>>(Qb, m_skip, XCb, UP, Kb);
  gemm(Kb, INNERc, WdnT, X, X1, Dc, NN, INNERc, Dc);

  // ================= block 1: sLSTM + FFN =================
  ln_kernel<<<NN, 256, 0, stream>>>(X1, s_ln_w, Abuf, Dc, 0);
  conv_silu_kernel<<<(NN * Dc) / 256, 256, 0, stream>>>(Abuf, Dc, s_cw, s_cb, XCb, Dc, Dc);
  {
    const float* srcs[4] = {XCb, XCb, Abuf, Abuf};
    for (int g = 0; g < 4; ++g)
      for (int hd = 0; hd < NHs; ++hd)
        gemm(srcs[g] + hd * DHs, Dc,
             WgT + ((size_t)g * NHs + hd) * DHs * DHs,
             nullptr, UP + (size_t)g * 4 * Mf + hd * DHs, Dc, NN, DHs, DHs);
  }
  slstm_kernel<<<Bc * NHs, DHs, 0, stream>>>(UP, UP + 4 * Mf, UP + 8 * Mf, UP + 12 * Mf,
                                             R4, s_bb, Kb);
  mhln_kernel<<<NN * NHs, 256, 0, stream>>>(Kb, Dc, s_mhln, X2, Dc, X1, DHs, NHs);
  ln_kernel<<<NN, 256, 0, stream>>>(X2, s_ffnln, Abuf, Dc, 0);
  gemm(Abuf, Dc, Wup2T, nullptr, UP, 2 * FFUP, NN, Dc, 2 * FFUP);
  gelu_mul_kernel<<<(NN * FFUP + 255) / 256, 256, 0, stream>>>(UP, Qb);
  gemm(Qb, FFUP, Wdn2T, X2, X1, Dc, NN, FFUP, Dc);

  // ================= post norm, SELU, pool, heads =================
  ln_kernel<<<NN, 256, 0, stream>>>(X1, post_ln, Abuf, Dc, 1);
  {
    dim3 g(Dc / 256, Bc);
    pool_kernel<<<g, 256, 0, stream>>>(Abuf, FEAT);
  }
  heads_kernel<<<Bc * 10, 256, 0, stream>>>(FEAT, h_We, h_be, h_Ws, h_bs, OUT);
}
